// Model_7816840479035
// MI455X (gfx1250) — matrix operand order repaired and checked by reading, not compiled, not run on hardware
//
#include <hip/hip_runtime.h>
#include <hip/hip_bf16.h>
#include <math.h>

// ---------------- model dims ----------------
#define Bb 8
#define Ss 4
#define Nn 2000
#define Tt 16
#define Ff 10
#define Ee 16
#define Rr 8
#define H1 16
#define H2 8
#define NHh 4
#define DKk 8
#define DVv 8
#define DMm (H2*Nn)   /* 16000 */
#define Dd (Rr*Tt)    /* 128 */
#define MLPu 16
#define SLOPE 0.1f

typedef __attribute__((ext_vector_type(16))) __bf16 bf16x16;
typedef __attribute__((ext_vector_type(8)))  float  f32x8;

static __device__ __forceinline__ float sigm(float x) { return 1.0f / (1.0f + __expf(-x)); }

// =====================================================================
// f32 -> bf16 conversion
// =====================================================================
__global__ void k_f32_to_bf16(const float* __restrict__ in, __bf16* __restrict__ out, long long n) {
    long long i = (long long)blockIdx.x * blockDim.x + threadIdx.x;
    if (i < n) out[i] = (__bf16)in[i];
}

// =====================================================================
// WMMA bf16 GEMM: C[M x Nc] = A[M x K] @ B[K x Nc]   (row-major, ld in elems)
// Tile: BM=128 x BN=64, BK=32. 256 threads = 8 waves (4 row x 2 col), each
// wave computes a 32x32 sub-tile via 2x2 v_wmma_f32_16x16x32_bf16.
// Interior tiles take a branch-free vectorized path (b64 global loads,
// b64 LDS stores) + global_prefetch of the next K-tile; edge tiles take
// the guarded scalar path.
// EPI: 0=plain, 1=acc*rowscale[row], 2=leaky_relu(acc)
// OUTBF: true -> store bf16, false -> store f32
// batching: z -> (i0 = z / batchDiv, i1 = z % batchDiv), element offsets.
// NOTE: requires lda/ldb even and >=8B-aligned base pointers (all call
// sites satisfy this: lds come from 256B-aligned workspace, lda in
// {2000,1024,128}).
// =====================================================================
#define BM 128
#define BN 64
#define BKc 32
#define LDA_P 40
#define LDB_P 40

template<int EPI, bool OUTBF>
__global__ __launch_bounds__(256)
void k_gemm_bf16(const __bf16* __restrict__ Ag, const __bf16* __restrict__ Bg,
                 void* __restrict__ Cg,
                 int M, int Nc, int K, int lda, int ldb, int ldc,
                 long long aS0, long long aS1, long long bS0, long long bS1,
                 long long cS0, long long cS1, int batchDiv,
                 const float* __restrict__ rowscale)
{
    __shared__ __bf16 As[BM * LDA_P];
    __shared__ __bf16 Bs[BN * LDB_P];

    const int z  = blockIdx.z;
    const int i0 = z / batchDiv, i1 = z % batchDiv;
    const __bf16* A  = Ag + i0 * aS0 + i1 * aS1;
    const __bf16* Bp = Bg + i0 * bS0 + i1 * bS1;

    const int rowBase = blockIdx.y * BM;
    const int colBase = blockIdx.x * BN;
    const int tid  = threadIdx.x;
    const int wave = tid >> 5;
    const int lane = tid & 31;
    const int lh   = lane & 15;
    const int sel  = lane >> 4;
    const int wm   = (wave & 3) << 5;   // wave row offset (0..96)
    const int wn   = (wave >> 2) << 5;  // wave col offset (0/32)

    const bool fullM = (rowBase + BM) <= M;

    // per-thread staging coordinates (constant across K-steps)
    const int arow = tid >> 3;              // + i*32 rows per iter  (4 iters)
    const int acol = (tid & 7) << 2;
    const int bkrow = tid >> 4;             // + i*16 rows per iter  (2 iters)
    const int bcol  = (tid & 15) << 2;

    union Frag  { bf16x16 v; unsigned u[8]; };
    union Pack4 { unsigned long long q; __bf16 h[4]; };

    f32x8 acc00 = {0.f,0.f,0.f,0.f,0.f,0.f,0.f,0.f};
    f32x8 acc01 = acc00, acc10 = acc00, acc11 = acc00;

    const int nk = (K + BKc - 1) / BKc;
    for (int kt = 0; kt < nk; ++kt) {
        const int k0 = kt * BKc;
        const bool fullK = (k0 + BKc) <= K;

        if (fullM && fullK) {
            // ---------- fast path: branch-free b64 loads/stores ----------
            #pragma unroll
            for (int i = 0; i < 4; ++i) {
                const int r = arow + i * 32;
                Pack4 p;
                p.q = *(const unsigned long long*)(A + (long long)(rowBase + r) * lda + k0 + acol);
                *(unsigned long long*)&As[r * LDA_P + acol] = p.q;
            }
            #pragma unroll
            for (int i = 0; i < 2; ++i) {
                const int kk = bkrow + i * 16;
                Pack4 p;
                p.q = *(const unsigned long long*)(Bp + (long long)(k0 + kk) * ldb + colBase + bcol);
                Bs[(bcol + 0) * LDB_P + kk] = p.h[0];
                Bs[(bcol + 1) * LDB_P + kk] = p.h[1];
                Bs[(bcol + 2) * LDB_P + kk] = p.h[2];
                Bs[(bcol + 3) * LDB_P + kk] = p.h[3];
            }
            // prefetch next K-tile (gfx1250 global_prefetch path)
            if (fullM && (k0 + 2 * BKc) <= K) {
                __builtin_prefetch(A + (long long)(rowBase + arow) * lda + k0 + BKc + acol, 0, 1);
                __builtin_prefetch(Bp + (long long)(k0 + BKc + bkrow) * ldb + colBase + bcol, 0, 1);
            }
        } else {
            // ---------- edge path: element-guarded ----------
            #pragma unroll
            for (int i = 0; i < 4; ++i) {
                const int r = arow + i * 32;
                const int gr = rowBase + r;
                const int gk = k0 + acol;
                __bf16 t0, t1, t2, t3;
                if (gr < M && (gk + 3) < K) {
                    const __bf16* p = A + (long long)gr * lda + gk;
                    t0 = p[0]; t1 = p[1]; t2 = p[2]; t3 = p[3];
                } else {
                    t0 = (gr < M && gk + 0 < K) ? A[(long long)gr * lda + gk + 0] : (__bf16)0.f;
                    t1 = (gr < M && gk + 1 < K) ? A[(long long)gr * lda + gk + 1] : (__bf16)0.f;
                    t2 = (gr < M && gk + 2 < K) ? A[(long long)gr * lda + gk + 2] : (__bf16)0.f;
                    t3 = (gr < M && gk + 3 < K) ? A[(long long)gr * lda + gk + 3] : (__bf16)0.f;
                }
                __bf16* dst = &As[r * LDA_P + acol];
                dst[0] = t0; dst[1] = t1; dst[2] = t2; dst[3] = t3;
            }
            #pragma unroll
            for (int i = 0; i < 2; ++i) {
                const int kk = bkrow + i * 16;
                const int gk = k0 + kk;
                __bf16 t0, t1, t2, t3;
                if (gk < K && (colBase + bcol + 3) < Nc) {
                    const __bf16* p = Bp + (long long)gk * ldb + colBase + bcol;
                    t0 = p[0]; t1 = p[1]; t2 = p[2]; t3 = p[3];
                } else {
                    t0 = (gk < K && colBase + bcol + 0 < Nc) ? Bp[(long long)gk * ldb + colBase + bcol + 0] : (__bf16)0.f;
                    t1 = (gk < K && colBase + bcol + 1 < Nc) ? Bp[(long long)gk * ldb + colBase + bcol + 1] : (__bf16)0.f;
                    t2 = (gk < K && colBase + bcol + 2 < Nc) ? Bp[(long long)gk * ldb + colBase + bcol + 2] : (__bf16)0.f;
                    t3 = (gk < K && colBase + bcol + 3 < Nc) ? Bp[(long long)gk * ldb + colBase + bcol + 3] : (__bf16)0.f;
                }
                Bs[(bcol + 0) * LDB_P + kk] = t0;
                Bs[(bcol + 1) * LDB_P + kk] = t1;
                Bs[(bcol + 2) * LDB_P + kk] = t2;
                Bs[(bcol + 3) * LDB_P + kk] = t3;
            }
        }
        __syncthreads();

        // ---- fragments (contiguous k = sel*16 + s in both A and B)
        Frag a0, a1, b0, b1;
        #pragma unroll
        for (int j = 0; j < 8; ++j) {
            const int kbA = sel * 16 + 2 * j;
            a0.u[j] = *(const unsigned*)&As[(wm +      lh) * LDA_P + kbA];
            a1.u[j] = *(const unsigned*)&As[(wm + 16 + lh) * LDA_P + kbA];
            const int kbB = sel * 16 + 2 * j;
            b0.u[j] = *(const unsigned*)&Bs[(wn +      lh) * LDB_P + kbB];
            b1.u[j] = *(const unsigned*)&Bs[(wn + 16 + lh) * LDB_P + kbB];
        }
        acc00 = __builtin_amdgcn_wmma_f32_16x16x32_bf16(false, a0.v, false, b0.v, (short)0, acc00, false, false);
        acc01 = __builtin_amdgcn_wmma_f32_16x16x32_bf16(false, a0.v, false, b1.v, (short)0, acc01, false, false);
        acc10 = __builtin_amdgcn_wmma_f32_16x16x32_bf16(false, a1.v, false, b0.v, (short)0, acc10, false, false);
        acc11 = __builtin_amdgcn_wmma_f32_16x16x32_bf16(false, a1.v, false, b1.v, (short)0, acc11, false, false);
        __syncthreads();
    }

    // ---- epilogue + store (C layout: lane 0-15 -> N=lane, row = v + sel*8)
    float*  Cf = (float*)Cg  + i0 * cS0 + i1 * cS1;
    __bf16* Cb = (__bf16*)Cg + i0 * cS0 + i1 * cS1;
    f32x8 accs[2][2] = {{acc00, acc01}, {acc10, acc11}};
    #pragma unroll
    for (int fi = 0; fi < 2; ++fi) {
        #pragma unroll
        for (int fj = 0; fj < 2; ++fj) {
            const int gc = colBase + wn + fj * 16 + lh;
            #pragma unroll
            for (int v = 0; v < 8; ++v) {
                const int gr = rowBase + wm + fi * 16 + v + sel * 8;
                if (gr < M && gc < Nc) {
                    float x = accs[fi][fj][v];
                    if (EPI == 1) x *= rowscale[gr];
                    if (EPI == 2) x = (x > 0.f) ? x : SLOPE * x;
                    if (OUTBF) Cb[(long long)gr * ldc + gc] = (__bf16)x;
                    else       Cf[(long long)gr * ldc + gc] = x;
                }
            }
        }
    }
}

// =====================================================================
// embed + LSTM1  (per (b,n) thread, scan over T)   h1: [B,N,T,H1]
// =====================================================================
__global__ __launch_bounds__(256)
void k_embed_lstm1(const float* __restrict__ in, const float* __restrict__ embW,
                   const float* __restrict__ embB, const float* __restrict__ W1,
                   const float* __restrict__ U1, const float* __restrict__ b1,
                   float* __restrict__ h1out)
{
    __shared__ float sW[Ee * 4 * H1];
    __shared__ float sU[H1 * 4 * H1];
    __shared__ float sE[Ff * Ee];
    __shared__ float sEb[Ee];
    __shared__ float sb[4 * H1];
    int tid = threadIdx.x;
    for (int i = tid; i < Ee * 4 * H1; i += 256) sW[i] = W1[i];
    for (int i = tid; i < H1 * 4 * H1; i += 256) sU[i] = U1[i];
    for (int i = tid; i < Ff * Ee; i += 256) sE[i] = embW[i];
    for (int i = tid; i < Ee; i += 256) sEb[i] = embB[i];
    for (int i = tid; i < 4 * H1; i += 256) sb[i] = b1[i];
    __syncthreads();

    int idx = blockIdx.x * 256 + tid;
    if (idx >= Bb * Nn) return;
    int b = idx / Nn, n = idx % Nn;
    const float* xin = in + (long long)(b * Nn + n) * Tt * Ff;
    float* outp = h1out + (long long)(b * Nn + n) * Tt * H1;

    float h[H1], c[H1];
    #pragma unroll
    for (int k2 = 0; k2 < H1; ++k2) { h[k2] = 0.f; c[k2] = 0.f; }

    for (int t = 0; t < Tt; ++t) {
        float xe[Ee];
        #pragma unroll
        for (int e = 0; e < Ee; ++e) {
            float a = sEb[e];
            #pragma unroll
            for (int f = 0; f < Ff; ++f) a += xin[t * Ff + f] * sE[f * Ee + e];
            xe[e] = a;
        }
        float hn[H1];
        #pragma unroll
        for (int k2 = 0; k2 < H1; ++k2) {
            float zi = sb[k2], zf = sb[H1 + k2], zg = sb[2 * H1 + k2], zo = sb[3 * H1 + k2];
            #pragma unroll
            for (int e = 0; e < Ee; ++e) {
                float xv = xe[e];
                zi += xv * sW[e * 64 + k2];
                zf += xv * sW[e * 64 + H1 + k2];
                zg += xv * sW[e * 64 + 2 * H1 + k2];
                zo += xv * sW[e * 64 + 3 * H1 + k2];
            }
            #pragma unroll
            for (int e = 0; e < H1; ++e) {
                float hv = h[e];
                zi += hv * sU[e * 64 + k2];
                zf += hv * sU[e * 64 + H1 + k2];
                zg += hv * sU[e * 64 + 2 * H1 + k2];
                zo += hv * sU[e * 64 + 3 * H1 + k2];
            }
            float cn = sigm(zf) * c[k2] + sigm(zi) * tanhf(zg);
            c[k2] = cn;
            hn[k2] = sigm(zo) * tanhf(cn);
        }
        #pragma unroll
        for (int k2 = 0; k2 < H1; ++k2) { h[k2] = hn[k2]; outp[t * H1 + k2] = hn[k2]; }
    }
}

// =====================================================================
// LayerNorm1 over (n,h) for each (b,t); h1 layout [B,N,T,H1], in-place
// =====================================================================
__global__ __launch_bounds__(256)
void k_ln1_stats(const float* __restrict__ h1, float* __restrict__ stats)
{
    int bt = blockIdx.x, b = bt / Tt, t = bt % Tt;
    float s = 0.f, ss = 0.f;
    for (int i = threadIdx.x; i < Nn * H1; i += 256) {
        int n = i >> 4, h = i & 15;
        float v = h1[((long long)(b * Nn + n) * Tt + t) * H1 + h];
        s += v; ss += v * v;
    }
    __shared__ float rs[256], rq[256];
    rs[threadIdx.x] = s; rq[threadIdx.x] = ss;
    __syncthreads();
    for (int o = 128; o > 0; o >>= 1) {
        if (threadIdx.x < o) { rs[threadIdx.x] += rs[threadIdx.x + o]; rq[threadIdx.x] += rq[threadIdx.x + o]; }
        __syncthreads();
    }
    if (threadIdx.x == 0) {
        float inv = 1.0f / (float)(Nn * H1);
        float mu = rs[0] * inv;
        float var = rq[0] * inv - mu * mu;
        stats[bt * 2] = mu;
        stats[bt * 2 + 1] = rsqrtf(var + 1e-5f);
    }
}

__global__ void k_ln1_apply(float* __restrict__ h1, const float* __restrict__ stats,
                            const float* __restrict__ g, const float* __restrict__ bta)
{
    long long i = (long long)blockIdx.x * 256 + threadIdx.x;
    if (i >= (long long)Bb * Nn * Tt * H1) return;
    int h = (int)(i & 15);
    long long r = i >> 4;
    int t = (int)(r % Tt); r /= Tt;
    int n = (int)(r % Nn); int b = (int)(r / Nn);
    float mu = stats[(b * Tt + t) * 2], rv = stats[(b * Tt + t) * 2 + 1];
    int feat = n * H1 + h;
    h1[i] = (h1[i] - mu) * rv * g[feat] + bta[feat];
}

// =====================================================================
// LSTM2: reads normalized h1 [B,N,T,16]; writes enc [B,T,N,H2]
// =====================================================================
__global__ __launch_bounds__(256)
void k_lstm2(const float* __restrict__ xin0, const float* __restrict__ W2,
             const float* __restrict__ U2, const float* __restrict__ b2,
             float* __restrict__ enc)
{
    __shared__ float sW[H1 * 4 * H2];   // 16*32
    __shared__ float sU[H2 * 4 * H2];   // 8*32
    __shared__ float sb[4 * H2];
    int tid = threadIdx.x;
    for (int i = tid; i < H1 * 4 * H2; i += 256) sW[i] = W2[i];
    for (int i = tid; i < H2 * 4 * H2; i += 256) sU[i] = U2[i];
    for (int i = tid; i < 4 * H2; i += 256) sb[i] = b2[i];
    __syncthreads();

    int idx = blockIdx.x * 256 + tid;
    if (idx >= Bb * Nn) return;
    int b = idx / Nn, n = idx % Nn;
    const float* xin = xin0 + (long long)(b * Nn + n) * Tt * H1;

    float h[H2], c[H2];
    #pragma unroll
    for (int k2 = 0; k2 < H2; ++k2) { h[k2] = 0.f; c[k2] = 0.f; }

    for (int t = 0; t < Tt; ++t) {
        float hn[H2];
        #pragma unroll
        for (int k2 = 0; k2 < H2; ++k2) {
            float zi = sb[k2], zf = sb[H2 + k2], zg = sb[2 * H2 + k2], zo = sb[3 * H2 + k2];
            #pragma unroll
            for (int e = 0; e < H1; ++e) {
                float xv = xin[t * H1 + e];
                zi += xv * sW[e * 32 + k2];
                zf += xv * sW[e * 32 + H2 + k2];
                zg += xv * sW[e * 32 + 2 * H2 + k2];
                zo += xv * sW[e * 32 + 3 * H2 + k2];
            }
            #pragma unroll
            for (int e = 0; e < H2; ++e) {
                float hv = h[e];
                zi += hv * sU[e * 32 + k2];
                zf += hv * sU[e * 32 + H2 + k2];
                zg += hv * sU[e * 32 + 2 * H2 + k2];
                zo += hv * sU[e * 32 + 3 * H2 + k2];
            }
            float cn = sigm(zf) * c[k2] + sigm(zi) * tanhf(zg);
            c[k2] = cn;
            hn[k2] = sigm(zo) * tanhf(cn);
        }
        #pragma unroll
        for (int k2 = 0; k2 < H2; ++k2) {
            h[k2] = hn[k2];
            enc[((long long)(b * Tt + t) * Nn + n) * H2 + k2] = hn[k2];
        }
    }
}

// =====================================================================
// q/k/v projections: per (b,t) block; q,k,v stored [B,T,32] (c = head*8+d)
// =====================================================================
__global__ __launch_bounds__(128)
void k_qkv(const float* __restrict__ enc, const float* __restrict__ Wq,
           const float* __restrict__ Wk, const float* __restrict__ Wv,
           float* __restrict__ q, float* __restrict__ k, float* __restrict__ v)
{
    __shared__ float row[DMm];
    int bt = blockIdx.x;
    const float* er = enc + (long long)bt * DMm;
    for (int j = threadIdx.x; j < DMm; j += 128) row[j] = er[j];
    __syncthreads();
    int tid = threadIdx.x;
    if (tid < 96) {
        int which = tid / 32, c = tid % 32;
        const float* W = (which == 0) ? Wq : ((which == 1) ? Wk : Wv);
        float a = 0.f;
        for (int j = 0; j < DMm; ++j) a += row[j] * W[j * 32 + c];
        float* dst = (which == 0) ? q : ((which == 1) ? k : v);
        dst[bt * 32 + c] = a;
    }
}

// =====================================================================
// causal attention core: one wave per (b, head); ctx [B,T,32]
// =====================================================================
__global__ void k_attn(const float* __restrict__ q, const float* __restrict__ kk,
                       const float* __restrict__ vv, float* __restrict__ ctx)
{
    int bh = blockIdx.x, b = bh / NHh, h = bh % NHh;
    __shared__ float ks[Tt][DKk], vs[Tt][DVv];
    for (int i = threadIdx.x; i < Tt * DKk; i += 32) {
        int tt = i / DKk, d = i % DKk;
        ks[tt][d] = kk[(b * Tt + tt) * 32 + h * DKk + d];
        vs[tt][d] = vv[(b * Tt + tt) * 32 + h * DVv + d];
    }
    __syncthreads();
    int t = threadIdx.x;
    if (t < Tt) {
        float qr[DKk];
        #pragma unroll
        for (int d = 0; d < DKk; ++d) qr[d] = q[(b * Tt + t) * 32 + h * DKk + d];
        float att[Tt];
        float mx = -1e30f;
        for (int tk = 0; tk <= t; ++tk) {
            float a = 0.f;
            #pragma unroll
            for (int d = 0; d < DKk; ++d) a += qr[d] * ks[tk][d];
            a *= 0.35355339059327379f;   // 1/sqrt(8)
            att[tk] = a;
            mx = fmaxf(mx, a);
        }
        float sum = 0.f;
        for (int tk = 0; tk <= t; ++tk) { att[tk] = __expf(att[tk] - mx); sum += att[tk]; }
        float inv = 1.0f / sum;
        float cr[DVv];
        #pragma unroll
        for (int d = 0; d < DVv; ++d) cr[d] = 0.f;
        for (int tk = 0; tk <= t; ++tk)
            #pragma unroll
            for (int d = 0; d < DVv; ++d) cr[d] += att[tk] * vs[tk][d];
        #pragma unroll
        for (int d = 0; d < DVv; ++d) ctx[(b * Tt + t) * 32 + h * DVv + d] = cr[d] * inv;
    }
}

// =====================================================================
// enc += ctx @ Wo  (residual), in place on enc [B,T,DM]
// =====================================================================
__global__ __launch_bounds__(256)
void k_resid_wo(float* __restrict__ enc, const float* __restrict__ ctx,
                const float* __restrict__ Wo)
{
    int bt = blockIdx.y;
    int j = blockIdx.x * 256 + threadIdx.x;
    __shared__ float cs[32];
    if (threadIdx.x < 32) cs[threadIdx.x] = ctx[bt * 32 + threadIdx.x];
    __syncthreads();
    if (j < DMm) {
        long long idx = (long long)bt * DMm + j;
        float a = enc[idx];
        #pragma unroll
        for (int c = 0; c < 32; ++c) a += cs[c] * Wo[(long long)c * DMm + j];
        enc[idx] = a;
    }
}

// =====================================================================
// LayerNorm2 over DM per (b,t), contiguous rows of enc, in-place
// =====================================================================
__global__ __launch_bounds__(256)
void k_ln2_stats(const float* __restrict__ enc, float* __restrict__ stats)
{
    int bt = blockIdx.x;
    const float* er = enc + (long long)bt * DMm;
    float s = 0.f, ss = 0.f;
    for (int i = threadIdx.x; i < DMm; i += 256) { float v = er[i]; s += v; ss += v * v; }
    __shared__ float rs[256], rq[256];
    rs[threadIdx.x] = s; rq[threadIdx.x] = ss;
    __syncthreads();
    for (int o = 128; o > 0; o >>= 1) {
        if (threadIdx.x < o) { rs[threadIdx.x] += rs[threadIdx.x + o]; rq[threadIdx.x] += rq[threadIdx.x + o]; }
        __syncthreads();
    }
    if (threadIdx.x == 0) {
        float inv = 1.0f / (float)DMm;
        float mu = rs[0] * inv;
        float var = rq[0] * inv - mu * mu;
        stats[bt * 2] = mu;
        stats[bt * 2 + 1] = rsqrtf(var + 1e-5f);
    }
}

__global__ void k_ln2_apply(float* __restrict__ enc, const float* __restrict__ stats,
                            const float* __restrict__ g, const float* __restrict__ bta)
{
    long long i = (long long)blockIdx.x * 256 + threadIdx.x;
    if (i >= (long long)Bb * Tt * DMm) return;
    int bt = (int)(i / DMm);
    int j  = (int)(i % DMm);
    float mu = stats[bt * 2], rv = stats[bt * 2 + 1];
    enc[i] = (enc[i] - mu) * rv * g[j] + bta[j];
}

// =====================================================================
// Build Xf in bf16, layout [n][b*128 + t*8 + h], from enc [B,T,N,H2]
// =====================================================================
__global__ void k_build_xf(const float* __restrict__ enc, __bf16* __restrict__ xf)
{
    int i = blockIdx.x * 256 + threadIdx.x;
    if (i >= Nn * 1024) return;
    int c = i & 1023, n = i >> 10;
    int b = c >> 7, r = c & 127, t = r >> 3, h = r & 7;
    xf[i] = (__bf16)enc[((long long)(b * Tt + t) * Nn + n) * H2 + h];
}

// =====================================================================
// hyper[n,c] = sum_s par[s] * F2[s][n][c]    (deterministic reduce)
// =====================================================================
__global__ void k_hyper_reduce(const float* __restrict__ F2, const float* __restrict__ par,
                               float* __restrict__ hyper)
{
    int i = blockIdx.x * 256 + threadIdx.x;
    if (i >= Nn * 1024) return;
    float a = 0.f;
    #pragma unroll
    for (int s = 0; s < Ss; ++s) a += par[s] * F2[(long long)s * (Nn * 1024) + i];
    hyper[i] = a;
}

// =====================================================================
// Final MLP: cat = [enc(b,n,t,:8), hyper(b,n,t,:8)] -> relu(cat@W1+b1)@W2+b2
// out [B,N,T]
// =====================================================================
__global__ __launch_bounds__(256)
void k_final_mlp(const float* __restrict__ enc, const float* __restrict__ hyper,
                 const float* __restrict__ W1m, const float* __restrict__ b1m,
                 const float* __restrict__ W2m, const float* __restrict__ b2m,
                 float* __restrict__ out)
{
    __shared__ float sW1[16 * MLPu], sb1[MLPu], sW2[MLPu];
    int tid = threadIdx.x;
    for (int i = tid; i < 16 * MLPu; i += 256) sW1[i] = W1m[i];
    for (int i = tid; i < MLPu; i += 256) { sb1[i] = b1m[i]; sW2[i] = W2m[i]; }
    __syncthreads();
    int i = blockIdx.x * 256 + tid;
    if (i >= Bb * Nn * Tt) return;
    int t = i % Tt, n = (i / Tt) % Nn, b = i / (Tt * Nn);
    float cat[16];
    #pragma unroll
    for (int h = 0; h < H2; ++h) {
        cat[h]     = enc[((long long)(b * Tt + t) * Nn + n) * H2 + h];
        cat[8 + h] = hyper[(long long)n * 1024 + b * 128 + t * 8 + h];
    }
    float acc = b2m[0];
    #pragma unroll
    for (int j = 0; j < MLPu; ++j) {
        float m = sb1[j];
        #pragma unroll
        for (int ii = 0; ii < 16; ++ii) m += cat[ii] * sW1[ii * MLPu + j];
        m = fmaxf(m, 0.f);
        acc += m * sW2[j];
    }
    out[i] = acc;
}

// =====================================================================
// host launcher
// =====================================================================
extern "C" void kernel_launch(void* const* d_in, const int* in_sizes, int n_in,
                              void* d_out, int out_size, void* d_ws, size_t ws_size,
                              hipStream_t stream)
{
    const float* inputs  = (const float*)d_in[0];
    const float* emb_W   = (const float*)d_in[1];
    const float* emb_b   = (const float*)d_in[2];
    const float* W1      = (const float*)d_in[3];
    const float* U1      = (const float*)d_in[4];
    const float* b1      = (const float*)d_in[5];
    const float* ln1_g   = (const float*)d_in[6];
    const float* ln1_b   = (const float*)d_in[7];
    const float* W2      = (const float*)d_in[8];
    const float* U2      = (const float*)d_in[9];
    const float* b2      = (const float*)d_in[10];
    const float* Wq      = (const float*)d_in[11];
    const float* Wk      = (const float*)d_in[12];
    const float* Wv      = (const float*)d_in[13];
    const float* Wo      = (const float*)d_in[14];
    const float* ln2_g   = (const float*)d_in[15];
    const float* ln2_b   = (const float*)d_in[16];
    const float* par     = (const float*)d_in[17];
    const float* theta1  = (const float*)d_in[18];
    const float* diag1   = (const float*)d_in[19];
    const float* theta2  = (const float*)d_in[20];
    const float* diag2   = (const float*)d_in[21];
    const float* mlp1_W  = (const float*)d_in[22];
    const float* mlp1_b  = (const float*)d_in[23];
    const float* mlp2_W  = (const float*)d_in[24];
    const float* mlp2_b  = (const float*)d_in[25];
    const float* wav     = (const float*)d_in[26];
    const float* wav_inv = (const float*)d_in[27];

    char* ws = (char*)d_ws;
    size_t off = 0;
    auto alloc = [&](size_t bytes) -> char* {
        char* p = ws + off;
        off = (off + bytes + 255) & ~(size_t)255;
        return p;
    };

    const long long NN  = (long long)Nn * Nn;       // 4,000,000 per snapshot
    const long long CW  = (long long)Nn * 1024;     // 2,048,000 (activation matrix)

    float*  h1     = (float*)alloc((size_t)Bb * Nn * Tt * H1 * 4);  // 16.4 MB
    float*  enc    = (float*)alloc((size_t)Bb * Tt * Nn * H2 * 4);  //  8.2 MB
    float*  stats1 = (float*)alloc(Bb * Tt * 2 * 4);
    float*  stats2 = (float*)alloc(Bb * Tt * 2 * 4);
    float*  qb     = (float*)alloc(Bb * Tt * 32 * 4);
    float*  kb     = (float*)alloc(Bb * Tt * 32 * 4);
    float*  vb     = (float*)alloc(Bb * Tt * 32 * 4);
    float*  ctxb   = (float*)alloc(Bb * Tt * 32 * 4);
    __bf16* Wbf    = (__bf16*)alloc((size_t)Ss * NN * 2);           // 32 MB
    __bf16* Wibf   = (__bf16*)alloc((size_t)Ss * NN * 2);           // 32 MB
    __bf16* th1b   = (__bf16*)alloc((size_t)Dd * Dd * 2);
    __bf16* th2b   = (__bf16*)alloc((size_t)Dd * Dd * 2);
    __bf16* xfb    = (__bf16*)alloc((size_t)CW * 2);                //  4 MB
    __bf16* x1b    = (__bf16*)alloc((size_t)CW * 2);                //  4 MB
    __bf16* t1b    = (__bf16*)alloc((size_t)Ss * CW * 2);           // 16 MB (also reused for X2)
    __bf16* f1b    = (__bf16*)alloc((size_t)Ss * CW * 2);           // 16 MB (also reused for t2)
    float*  F2     = (float*)alloc((size_t)Ss * CW * 4);            // 32 MB
    float*  hyper  = (float*)alloc((size_t)CW * 4);                 //  8 MB
    (void)ws_size; (void)in_sizes; (void)n_in; (void)out_size;

    // --- precision conversion (once per call; inputs untouched)
    {
        long long n = (long long)Ss * NN;
        int g = (int)((n + 255) / 256);
        k_f32_to_bf16<<<g, 256, 0, stream>>>(wav, Wbf, n);
        k_f32_to_bf16<<<g, 256, 0, stream>>>(wav_inv, Wibf, n);
        k_f32_to_bf16<<<(Dd * Dd + 255) / 256, 256, 0, stream>>>(theta1, th1b, Dd * Dd);
        k_f32_to_bf16<<<(Dd * Dd + 255) / 256, 256, 0, stream>>>(theta2, th2b, Dd * Dd);
    }

    // --- encoder
    k_embed_lstm1<<<(Bb * Nn + 255) / 256, 256, 0, stream>>>(inputs, emb_W, emb_b, W1, U1, b1, h1);
    k_ln1_stats<<<Bb * Tt, 256, 0, stream>>>(h1, stats1);
    k_ln1_apply<<<(int)(((long long)Bb * Nn * Tt * H1 + 255) / 256), 256, 0, stream>>>(h1, stats1, ln1_g, ln1_b);
    k_lstm2<<<(Bb * Nn + 255) / 256, 256, 0, stream>>>(h1, W2, U2, b2, enc);

    // --- temporal attention + residual + LN
    k_qkv<<<Bb * Tt, 128, 0, stream>>>(enc, Wq, Wk, Wv, qb, kb, vb);
    k_attn<<<Bb * NHh, 32, 0, stream>>>(qb, kb, vb, ctxb);
    k_resid_wo<<<dim3((DMm + 255) / 256, Bb * Tt), 256, 0, stream>>>(enc, ctxb, Wo);
    k_ln2_stats<<<Bb * Tt, 256, 0, stream>>>(enc, stats2);
    k_ln2_apply<<<(int)(((long long)Bb * Tt * DMm + 255) / 256), 256, 0, stream>>>(enc, stats2, ln2_g, ln2_b);

    // --- hypergraph wavelet conv (WMMA bf16 GEMM chain)
    k_build_xf<<<(int)((CW + 255) / 256), 256, 0, stream>>>(enc, xfb);

    // X1 = Xf @ theta1   (per-b column block; z = b, batchDiv=1)
    k_gemm_bf16<0, true><<<dim3(2, 16, Bb), 256, 0, stream>>>(
        xfb, th1b, x1b, Nn, 128, 128, 1024, 128, 1024,
        128, 0, 0, 0, 128, 0, 1, nullptr);

    // t1 = (Winv[s] @ X1) * diag1[n]
    k_gemm_bf16<1, true><<<dim3(16, 16, Ss), 256, 0, stream>>>(
        Wibf, x1b, t1b, Nn, 1024, Nn, Nn, 1024, 1024,
        NN, 0, 0, 0, CW, 0, 1, diag1);

    // f1 = leaky_relu(W[s] @ t1)
    k_gemm_bf16<2, true><<<dim3(16, 16, Ss), 256, 0, stream>>>(
        Wbf, t1b, f1b, Nn, 1024, Nn, Nn, 1024, 1024,
        NN, 0, CW, 0, CW, 0, 1, nullptr);

    // X2 = f1 @ theta2   (z -> (s,b); batchDiv=8) ; overwrite t1b
    k_gemm_bf16<0, true><<<dim3(2, 16, Ss * Bb), 256, 0, stream>>>(
        f1b, th2b, t1b, Nn, 128, 128, 1024, 128, 1024,
        CW, 128, 0, 0, CW, 128, Bb, nullptr);

    // t2 = (Winv[s] @ X2) * diag2[n]  ; overwrite f1b
    k_gemm_bf16<1, true><<<dim3(16, 16, Ss), 256, 0, stream>>>(
        Wibf, t1b, f1b, Nn, 1024, Nn, Nn, 1024, 1024,
        NN, 0, CW, 0, CW, 0, 1, diag2);

    // f2 = leaky_relu(W[s] @ t2) -> F2 (f32)
    k_gemm_bf16<2, false><<<dim3(16, 16, Ss), 256, 0, stream>>>(
        Wbf, f1b, F2, Nn, 1024, Nn, Nn, 1024, 1024,
        NN, 0, CW, 0, CW, 0, 1, nullptr);

    // hyper = sum_s par[s] * f2[s]
    k_hyper_reduce<<<(int)((CW + 255) / 256), 256, 0, stream>>>(F2, par, hyper);

    // --- final MLP
    k_final_mlp<<<(Bb * Nn * Tt + 255) / 256, 256, 0, stream>>>(
        enc, hyper, mlp1_W, mlp1_b, mlp2_W, mlp2_b, (float*)d_out);
}
